// Decoder_14044543057922
// MI455X (gfx1250) — matrix operand order repaired and checked by reading, not compiled, not run on hardware
//
#include <hip/hip_runtime.h>
#include <hip/hip_bf16.h>
#include <math.h>

// Problem dims (compile-time)
#define B_DIM 32
#define S_DIM 128
#define H_DIM 512
#define E_DIM 512
#define V_DIM 32000
#define G3H   1536   // 3*H
#define MROWS 4096   // S*B == B*S

typedef __bf16 bf16_t;
typedef __attribute__((ext_vector_type(16))) __bf16 v16bf;
typedef __attribute__((ext_vector_type(8)))  float  v8f;

union FragBF { v16bf v; uint4 q[2]; };

// A-operand fragment (16x32 bf16, M x K), memory row-major with leading dim lda.
// ISA layout: lane = m + 16*half; elements 0-7 = K[half*8 .. half*8+7],
// elements 8-15 = K[16+half*8 .. 16+half*8+7]  -> two contiguous 16B loads.
__device__ __forceinline__ v16bf load_frag_a(const bf16_t* __restrict__ base, int lda) {
  const int lane = threadIdx.x & 31;
  const int m  = lane & 15;
  const int hh = lane >> 4;
  const bf16_t* p = base + (size_t)m * lda + hh * 8;
  FragBF f;
  f.q[0] = *reinterpret_cast<const uint4*>(p);
  f.q[1] = *reinterpret_cast<const uint4*>(p + 16);
  return f.v;
}

// B-operand fragment (32x16 bf16, K x N). Weight matrices are stored row-major
// [N, K] (each output column n is a contiguous K-row), which matches the
// B layout: lane = n + 16*half; elements 0-7 = K[half*8 .. half*8+7], 8-15 = K[16+half*8 ..].
__device__ __forceinline__ v16bf load_frag_b(const bf16_t* __restrict__ base, int ldb) {
  const int lane = threadIdx.x & 31;
  const int n  = lane & 15;
  const int hh = lane >> 4;
  const bf16_t* p = base + (size_t)n * ldb + hh * 8;
  FragBF f;
  f.q[0] = *reinterpret_cast<const uint4*>(p);
  f.q[1] = *reinterpret_cast<const uint4*>(p + 16);
  return f.v;
}

__device__ __forceinline__ v8f wmma_bf16(v16bf a, v16bf b, v8f c) {
  return __builtin_amdgcn_wmma_f32_16x16x32_bf16(
      /*neg_a=*/false, a, /*neg_b=*/false, b,
      /*c_mod=*/(short)0, c, /*reuse_a=*/false, /*reuse_b=*/false);
}

// ---------------------------------------------------------------------------
// Generic GEMM:  C[m,n] = sum_k A[m,k]*Bw[n,k] + bias[n]
// A: [M,K] bf16 row-major; Bw: [N,K] bf16 row-major; C: [M,N] f32 row-major.
// Block = 8 waves in a 2(M) x 4(N) grid; wave tile = 64x64 -> block 128x256.
// Per-k working set (8KB A + 16KB B) fits WGP$, so intra-block reuse is in L0.
// Output stores are non-temporal so the 524MB logits stream doesn't evict the
// L2-resident bf16 weights.
// Requires M % 128 == 0 and N % 256 == 0 (holds for both call sites).
// ---------------------------------------------------------------------------
__global__ void __launch_bounds__(256) gemm_bf16_wmma(
    const bf16_t* __restrict__ A, const bf16_t* __restrict__ Bw,
    const float* __restrict__ bias, float* __restrict__ C,
    int M, int N, int K) {
  const int wave = threadIdx.x >> 5;
  const int wm = wave >> 2;  // 0..1
  const int wn = wave & 3;   // 0..3
  const int m0 = blockIdx.y * 128 + wm * 64;
  const int n0 = blockIdx.x * 256 + wn * 64;

  v8f acc[4][4];
#pragma unroll
  for (int t = 0; t < 4; ++t)
#pragma unroll
    for (int u = 0; u < 4; ++u) acc[t][u] = (v8f)0.0f;

  const int kiters = K >> 5;
  for (int kk = 0; kk < kiters; ++kk) {
    const int kof = kk * 32;
    v16bf a[4];
#pragma unroll
    for (int t = 0; t < 4; ++t)
      a[t] = load_frag_a(A + (size_t)(m0 + t * 16) * K + kof, K);
#pragma unroll
    for (int u = 0; u < 4; ++u) {
      v16bf b = load_frag_b(Bw + (size_t)(n0 + u * 16) * K + kof, K);
#pragma unroll
      for (int t = 0; t < 4; ++t) acc[t][u] = wmma_bf16(a[t], b, acc[t][u]);
    }
  }

  const int lane = threadIdx.x & 31;
  const int nl = lane & 15;
  const int hh = lane >> 4;
#pragma unroll
  for (int u = 0; u < 4; ++u) {
    const int ncol = n0 + u * 16 + nl;
    const float bv = bias[ncol];
#pragma unroll
    for (int t = 0; t < 4; ++t) {
#pragma unroll
      for (int r = 0; r < 8; ++r) {
        const int row = m0 + t * 16 + hh * 8 + r;
        __builtin_nontemporal_store(acc[t][u][r] + bv, &C[(size_t)row * N + ncol]);
      }
    }
  }
}

// ---------------------------------------------------------------------------
// One GRU time step.  4 blocks x 8 waves; wave owns 16 hidden columns j,
// computes gh tiles for all three gates (cols j, H+j, 2H+j), then the
// nonlinearities and the new hidden state (f32 + bf16 copies + xs for GEMM2).
// ---------------------------------------------------------------------------
__global__ void __launch_bounds__(256) gru_step(
    const bf16_t* __restrict__ h_bf_in, const float* __restrict__ h_f32_in,
    const bf16_t* __restrict__ Whh_b,   const float* __restrict__ b_hh,
    const float* __restrict__ xgates,
    bf16_t* __restrict__ h_bf_out, float* __restrict__ h_f32_out,
    bf16_t* __restrict__ xs, int s) {
  const int wave = threadIdx.x >> 5;
  const int j16 = blockIdx.x * 128 + wave * 16;

  v8f acc[2][3];
#pragma unroll
  for (int t = 0; t < 2; ++t)
#pragma unroll
    for (int g = 0; g < 3; ++g) acc[t][g] = (v8f)0.0f;

  for (int kk = 0; kk < (H_DIM >> 5); ++kk) {
    const int kof = kk * 32;
    v16bf a0 = load_frag_a(h_bf_in + kof, H_DIM);                      // batch 0..15
    v16bf a1 = load_frag_a(h_bf_in + (size_t)16 * H_DIM + kof, H_DIM); // batch 16..31
#pragma unroll
    for (int g = 0; g < 3; ++g) {
      v16bf b = load_frag_b(Whh_b + (size_t)(g * H_DIM + j16) * H_DIM + kof, H_DIM);
      acc[0][g] = wmma_bf16(a0, b, acc[0][g]);
      acc[1][g] = wmma_bf16(a1, b, acc[1][g]);
    }
  }

  const int lane = threadIdx.x & 31;
  const int nl = lane & 15;
  const int hh = lane >> 4;
  const int j = j16 + nl;
  const float bhr = b_hh[j];
  const float bhz = b_hh[H_DIM + j];
  const float bhn = b_hh[2 * H_DIM + j];

#pragma unroll
  for (int t = 0; t < 2; ++t) {
#pragma unroll
    for (int r = 0; r < 8; ++r) {
      const int bi = t * 16 + hh * 8 + r;  // batch index
      const float* xg = xgates + ((size_t)s * B_DIM + bi) * G3H + j;
      const float ghr = acc[t][0][r] + bhr;
      const float ghz = acc[t][1][r] + bhz;
      const float ghn = acc[t][2][r] + bhn;
      const float rg = 1.0f / (1.0f + expf(-(xg[0] + ghr)));
      const float zg = 1.0f / (1.0f + expf(-(xg[H_DIM] + ghz)));
      const float ng = tanhf(xg[2 * H_DIM] + rg * ghn);
      const float hold = h_f32_in[(size_t)bi * H_DIM + j];
      const float hn = (1.0f - zg) * ng + zg * hold;
      h_f32_out[(size_t)bi * H_DIM + j] = hn;
      h_bf_out[(size_t)bi * H_DIM + j] = (bf16_t)hn;
      // store transposed to [B, S, H] for the output GEMM (row = b*S + s)
      xs[((size_t)bi * S_DIM + s) * H_DIM + j] = (bf16_t)hn;
    }
  }
}

// ---------------------------------------------------------------------------
// Utility kernels
// ---------------------------------------------------------------------------
__global__ void __launch_bounds__(256) f32_to_bf16_kernel(
    const float* __restrict__ in, bf16_t* __restrict__ out, int n) {
  for (int i = blockIdx.x * 256 + threadIdx.x; i < n; i += gridDim.x * 256)
    out[i] = (bf16_t)in[i];
}

// Gather embedding rows for tok_in[b,s] = (s==0 ? init : targets[b,s-1]),
// row r = s*B + b, converting to bf16.
__global__ void __launch_bounds__(256) gather_embed_kernel(
    const float* __restrict__ embedding, const int* __restrict__ targets,
    const int* __restrict__ init_tok, bf16_t* __restrict__ X1) {
  const int r = blockIdx.x;           // 0..4095
  const int s = r >> 5;               // r / B
  const int b = r & 31;               // r % B
  const int tok = (s == 0) ? init_tok[0] : targets[b * S_DIM + (s - 1)];
  const float* src = embedding + (size_t)tok * E_DIM;
  bf16_t* dst = X1 + (size_t)r * E_DIM;
  for (int e = threadIdx.x; e < E_DIM; e += 256) dst[e] = (bf16_t)src[e];
}

__global__ void __launch_bounds__(256) init_h_kernel(
    const float* __restrict__ hidden, float* __restrict__ hf,
    bf16_t* __restrict__ hb) {
  const int i = blockIdx.x * 256 + threadIdx.x;
  if (i < B_DIM * H_DIM) {
    const float v = hidden[i];
    hf[i] = v;
    hb[i] = (bf16_t)v;
  }
}

// ---------------------------------------------------------------------------
extern "C" void kernel_launch(void* const* d_in, const int* in_sizes, int n_in,
                              void* d_out, int out_size, void* d_ws, size_t ws_size,
                              hipStream_t stream) {
  (void)in_sizes; (void)n_in; (void)out_size; (void)ws_size;

  const float* hidden    = (const float*)d_in[0];
  const int*   targets   = (const int*)d_in[1];
  const float* embedding = (const float*)d_in[2];
  const float* W_ih      = (const float*)d_in[3];
  const float* W_hh      = (const float*)d_in[4];
  const float* b_ih      = (const float*)d_in[5];
  const float* b_hh      = (const float*)d_in[6];
  const float* W_out     = (const float*)d_in[7];
  const float* b_out     = (const float*)d_in[8];
  const int*   init_tok  = (const int*)d_in[9];

  // Workspace carve-out (~70 MB total)
  char* ws = (char*)d_ws;
  size_t off = 0;
  auto take = [&](size_t bytes) -> void* {
    void* p = ws + off;
    off += (bytes + 255) & ~(size_t)255;
    return p;
  };
  bf16_t* Wih_b  = (bf16_t*)take((size_t)G3H * E_DIM * 2);
  bf16_t* Whh_b  = (bf16_t*)take((size_t)G3H * H_DIM * 2);
  bf16_t* Wout_b = (bf16_t*)take((size_t)V_DIM * H_DIM * 2);
  bf16_t* X1     = (bf16_t*)take((size_t)MROWS * E_DIM * 2);   // gathered embeddings
  float*  xgates = (float*)take((size_t)MROWS * G3H * 4);      // [S,B,3H]
  bf16_t* xs     = (bf16_t*)take((size_t)B_DIM * S_DIM * H_DIM * 2); // [B,S,H]
  float*  hf[2]; bf16_t* hb[2];
  hf[0] = (float*)take((size_t)B_DIM * H_DIM * 4);
  hf[1] = (float*)take((size_t)B_DIM * H_DIM * 4);
  hb[0] = (bf16_t*)take((size_t)B_DIM * H_DIM * 2);
  hb[1] = (bf16_t*)take((size_t)B_DIM * H_DIM * 2);

  // 1) Convert weights to bf16
  f32_to_bf16_kernel<<<1024, 256, 0, stream>>>(W_ih, Wih_b, G3H * E_DIM);
  f32_to_bf16_kernel<<<1024, 256, 0, stream>>>(W_hh, Whh_b, G3H * H_DIM);
  f32_to_bf16_kernel<<<4096, 256, 0, stream>>>(W_out, Wout_b, V_DIM * H_DIM);

  // 2) Gather used embedding rows (4096 rows only, not all 32000)
  gather_embed_kernel<<<MROWS, 256, 0, stream>>>(embedding, targets, init_tok, X1);

  // 3) x_gates = X1 @ W_ih^T + b_ih : [4096, 1536]
  {
    dim3 grid(G3H / 256, MROWS / 128);  // (6, 32), exact
    gemm_bf16_wmma<<<grid, 256, 0, stream>>>(X1, Wih_b, b_ih, xgates,
                                             MROWS, G3H, E_DIM);
  }

  // 4) Init hidden state (f32 + bf16 ping-pong buffers)
  init_h_kernel<<<(B_DIM * H_DIM + 255) / 256, 256, 0, stream>>>(hidden, hf[0], hb[0]);

  // 5) Sequential GRU scan: 128 small launches on the stream (graph-captured)
  for (int s = 0; s < S_DIM; ++s) {
    const int pi = s & 1, po = (s + 1) & 1;
    gru_step<<<4, 256, 0, stream>>>(hb[pi], hf[pi], Whh_b, b_hh, xgates,
                                    hb[po], hf[po], xs, s);
  }

  // 6) logits = xs @ W_out^T + b_out : [4096, 32000] -> d_out [B,S,V]
  {
    dim3 grid(V_DIM / 256, MROWS / 128);  // (125, 32), exact
    gemm_bf16_wmma<<<grid, 256, 0, stream>>>(xs, Wout_b, b_out, (float*)d_out,
                                             MROWS, V_DIM, H_DIM);
  }
}
